// NetTransform_38362647888184
// MI455X (gfx1250) — matrix operand order repaired and checked by reading, not compiled, not run on hardware
//
#include <hip/hip_runtime.h>
#include <hip/hip_bf16.h>

#define B_ 8
#define T_ 2048
#define H_ 512

typedef __attribute__((ext_vector_type(4)))  float  f32x4;
typedef __attribute__((ext_vector_type(8)))  float  f32x8;
typedef __attribute__((ext_vector_type(16))) __bf16 bf16x16;
typedef __attribute__((ext_vector_type(4)))  __bf16 bf16x4;

// ---------------------------------------------------------------------------
// Kernel 1: global min/max of e (B*T = 16384 floats)
// ---------------------------------------------------------------------------
__global__ __launch_bounds__(256)
void minmax_kernel(const float* __restrict__ e, float* __restrict__ mm) {
    __shared__ float smn[256];
    __shared__ float smx[256];
    const int tid = threadIdx.x;
    float mn = 3.402823466e38f, mx = -3.402823466e38f;
    for (int i = tid; i < B_ * T_; i += 256) {
        float v = e[i];
        mn = fminf(mn, v);
        mx = fmaxf(mx, v);
    }
    smn[tid] = mn; smx[tid] = mx;
    __syncthreads();
    for (int s = 128; s > 0; s >>= 1) {
        if (tid < s) {
            smn[tid] = fminf(smn[tid], smn[tid + s]);
            smx[tid] = fmaxf(smx[tid], smx[tid + s]);
        }
        __syncthreads();
    }
    if (tid == 0) { mm[0] = smn[0]; mm[1] = smx[0]; }
}

// ---------------------------------------------------------------------------
// Kernel 2: per-batch linear-recurrence scan.
//   x[t]    = (e[t]-mn)/(mx-mn)
//   Ksum[t] = 1 + x[t-1]*Ksum[t-1]
//   N[t,k]  = h[t,k] + x[t-1]*N[t-1,k]
//   Agg[t,k] = N[t,k] / Ksum[t]
// One block per batch, 512 threads (one per column k of H).
// ---------------------------------------------------------------------------
__global__ __launch_bounds__(512)
void agg_scan_kernel(const float* __restrict__ e, const float* __restrict__ h,
                     const float* __restrict__ mm, float* __restrict__ agg) {
    __shared__ float xs[T_];
    __shared__ float rk[T_];
    const int b   = blockIdx.x;
    const int tid = threadIdx.x;
    const float mn = mm[0];
    const float dnm = mm[1] - mn;
    for (int i = tid; i < T_; i += 512)
        xs[i] = (e[b * T_ + i] - mn) / dnm;
    __syncthreads();
    if (tid == 0) {
        float ks = 1.0f;
        rk[0] = 1.0f;
        for (int t = 1; t < T_; ++t) {
            ks = fmaf(xs[t - 1], ks, 1.0f);
            rk[t] = 1.0f / ks;
        }
    }
    __syncthreads();
    const float* hb = h   + (size_t)b * T_ * H_ + tid;
    float*       ab = agg + (size_t)b * T_ * H_ + tid;
    float n = 0.0f;
    float xprev = 0.0f;
    // Unroll so the 4 independent h loads issue ahead of the dependent chain.
    for (int t = 0; t < T_; t += 4) {
        float h0 = hb[(size_t)(t + 0) * H_];
        float h1 = hb[(size_t)(t + 1) * H_];
        float h2 = hb[(size_t)(t + 2) * H_];
        float h3 = hb[(size_t)(t + 3) * H_];
        n = fmaf(xprev, n, h0);  ab[(size_t)(t + 0) * H_] = n * rk[t + 0];
        n = fmaf(xs[t + 0], n, h1); ab[(size_t)(t + 1) * H_] = n * rk[t + 1];
        n = fmaf(xs[t + 1], n, h2); ab[(size_t)(t + 2) * H_] = n * rk[t + 2];
        n = fmaf(xs[t + 2], n, h3); ab[(size_t)(t + 3) * H_] = n * rk[t + 3];
        xprev = xs[t + 3];
    }
}

// ---------------------------------------------------------------------------
// Kernel 3: batched NT GEMM  out[b,i,j] = (1/sqrt(H)) * sum_k h[b,i,k]*Agg[b,j,k]
// Block tile 128(M) x 64(N); K chunks of 32; 8 waves of 32x32 each.
// fp32 operands split into (hi,lo) bf16; 3x v_wmma_f32_16x16x32_bf16 per
// 16x16x32 product gives ~fp32 accuracy at tensor-core rate.
// Double-buffered LDS (2 x 30 KB): next chunk's global loads issue before the
// current chunk's WMMAs, one barrier per iteration.
// LDS rows padded to 40 bf16 (80 B) -> conflict-free b128 fragment loads.
// ---------------------------------------------------------------------------
__global__ __launch_bounds__(256)
void gemm_kernel(const float* __restrict__ h, const float* __restrict__ agg,
                 float* __restrict__ out) {
    __shared__ __align__(16) __bf16 hHi[2][128 * 40];
    __shared__ __align__(16) __bf16 hLo[2][128 * 40];
    __shared__ __align__(16) __bf16 gHi[2][64 * 40];
    __shared__ __align__(16) __bf16 gLo[2][64 * 40];

    const int b     = blockIdx.z;
    const int mBase = blockIdx.x * 128;
    const int nBase = blockIdx.y * 64;
    const int tid   = threadIdx.x;
    const int lane  = tid & 31;
    const int wave  = tid >> 5;
    const int wm    = (wave & 3) * 32;   // wave M offset within block tile
    const int wn    = (wave >> 2) * 32;  // wave N offset within block tile
    const int lrow  = lane & 15;
    const int lhalf = lane >> 4;

    const float* hB = h   + (size_t)b * T_ * H_;
    const float* gB = agg + (size_t)b * T_ * H_;

    // ---- per-thread staging coordinates (K-loop invariant) ----
    const float* hp[4];
    int offH[4];
#pragma unroll
    for (int j = 0; j < 4; ++j) {
        int idx = tid + j * 256;            // 0..1023 float4 slots of 128x32
        int row = idx >> 3, c4 = idx & 7;
        hp[j]   = hB + (size_t)(mBase + row) * H_ + c4 * 4;
        offH[j] = row * 40 + c4 * 4;
    }
    const float* gp[2];
    int offG[2];
#pragma unroll
    for (int j = 0; j < 2; ++j) {
        int idx = tid + j * 256;            // 0..511 float4 slots of 64x32
        int row = idx >> 3, c4 = idx & 7;
        gp[j]   = gB + (size_t)(nBase + row) * H_ + c4 * 4;
        offG[j] = row * 40 + c4 * 4;
    }

    f32x8 acc[2][2] = {};
    union Frag { bf16x16 v; f32x4 q[2]; };

    f32x4 vh[4], vg[2];

    // ---- prologue: stage K-chunk 0 into buffer 0 ----
#pragma unroll
    for (int j = 0; j < 4; ++j) vh[j] = *(const f32x4*)(hp[j]);
#pragma unroll
    for (int j = 0; j < 2; ++j) vg[j] = *(const f32x4*)(gp[j]);
#pragma unroll
    for (int j = 0; j < 4; ++j) {
        bf16x4 hi, lo;
#pragma unroll
        for (int q = 0; q < 4; ++q) {
            __bf16 hq = (__bf16)vh[j][q];
            hi[q] = hq; lo[q] = (__bf16)(vh[j][q] - (float)hq);
        }
        *(bf16x4*)(&hHi[0][offH[j]]) = hi;
        *(bf16x4*)(&hLo[0][offH[j]]) = lo;
    }
#pragma unroll
    for (int j = 0; j < 2; ++j) {
        bf16x4 hi, lo;
#pragma unroll
        for (int q = 0; q < 4; ++q) {
            __bf16 hq = (__bf16)vg[j][q];
            hi[q] = hq; lo[q] = (__bf16)(vg[j][q] - (float)hq);
        }
        *(bf16x4*)(&gHi[0][offG[j]]) = hi;
        *(bf16x4*)(&gLo[0][offG[j]]) = lo;
    }
    __syncthreads();

    constexpr int KITER = H_ / 32;
#pragma unroll 2
    for (int it = 0; it < KITER; ++it) {
        const int cur = it & 1;
        const int nxt = cur ^ 1;

        // ---- issue next chunk's global loads early (hidden behind WMMAs) ----
        if (it + 1 < KITER) {
            const int k0 = (it + 1) * 32;
#pragma unroll
            for (int j = 0; j < 4; ++j) vh[j] = *(const f32x4*)(hp[j] + k0);
#pragma unroll
            for (int j = 0; j < 2; ++j) vg[j] = *(const f32x4*)(gp[j] + k0);
        }

        // ---- fragments (ISA 16-bit A 16x32 layout: K runs of 8 at
        //      {base, base+16}, base = lane_half*8; B: same K runs) ----
        Frag fAhi[2], fAlo[2], fBhi[2], fBlo[2];
#pragma unroll
        for (int ms = 0; ms < 2; ++ms) {
            int r  = wm + ms * 16 + lrow;
            int ks = lhalf * 8;
            fAhi[ms].q[0] = *(const f32x4*)(&hHi[cur][r * 40 + ks]);
            fAhi[ms].q[1] = *(const f32x4*)(&hHi[cur][r * 40 + ks + 16]);
            fAlo[ms].q[0] = *(const f32x4*)(&hLo[cur][r * 40 + ks]);
            fAlo[ms].q[1] = *(const f32x4*)(&hLo[cur][r * 40 + ks + 16]);
        }
#pragma unroll
        for (int ns = 0; ns < 2; ++ns) {
            int r  = wn + ns * 16 + lrow;
            int ks = lhalf * 8;
            fBhi[ns].q[0] = *(const f32x4*)(&gHi[cur][r * 40 + ks]);
            fBhi[ns].q[1] = *(const f32x4*)(&gHi[cur][r * 40 + ks + 16]);
            fBlo[ns].q[0] = *(const f32x4*)(&gLo[cur][r * 40 + ks]);
            fBlo[ns].q[1] = *(const f32x4*)(&gLo[cur][r * 40 + ks + 16]);
        }

        // ---- 3-term bf16-split products, fp32 accumulate ----
#pragma unroll
        for (int ms = 0; ms < 2; ++ms)
#pragma unroll
            for (int ns = 0; ns < 2; ++ns) {
                acc[ms][ns] = __builtin_amdgcn_wmma_f32_16x16x32_bf16(
                    false, fAhi[ms].v, false, fBhi[ns].v, (short)0, acc[ms][ns], false, false);
                acc[ms][ns] = __builtin_amdgcn_wmma_f32_16x16x32_bf16(
                    false, fAhi[ms].v, false, fBlo[ns].v, (short)0, acc[ms][ns], false, false);
                acc[ms][ns] = __builtin_amdgcn_wmma_f32_16x16x32_bf16(
                    false, fAlo[ms].v, false, fBhi[ns].v, (short)0, acc[ms][ns], false, false);
            }

        // ---- convert + store next chunk into the other buffer ----
        if (it + 1 < KITER) {
#pragma unroll
            for (int j = 0; j < 4; ++j) {
                bf16x4 hi, lo;
#pragma unroll
                for (int q = 0; q < 4; ++q) {
                    __bf16 hq = (__bf16)vh[j][q];
                    hi[q] = hq; lo[q] = (__bf16)(vh[j][q] - (float)hq);
                }
                *(bf16x4*)(&hHi[nxt][offH[j]]) = hi;
                *(bf16x4*)(&hLo[nxt][offH[j]]) = lo;
            }
#pragma unroll
            for (int j = 0; j < 2; ++j) {
                bf16x4 hi, lo;
#pragma unroll
                for (int q = 0; q < 4; ++q) {
                    __bf16 hq = (__bf16)vg[j][q];
                    hi[q] = hq; lo[q] = (__bf16)(vg[j][q] - (float)hq);
                }
                *(bf16x4*)(&gHi[nxt][offG[j]]) = hi;
                *(bf16x4*)(&gLo[nxt][offG[j]]) = lo;
            }
        }
        __syncthreads();
    }

    // ---- epilogue: C/D layout (VGPR v -> M = v + 8*lane_half, N = lane%16) ----
    const float scale = 0.04419417382415922f;  // 1/sqrt(512)
#pragma unroll
    for (int ms = 0; ms < 2; ++ms)
#pragma unroll
        for (int ns = 0; ns < 2; ++ns) {
            int col  = nBase + wn + ns * 16 + lrow;
            int rowB = mBase + wm + ms * 16 + (lhalf << 3);
#pragma unroll
            for (int v = 0; v < 8; ++v) {
                out[(size_t)b * T_ * T_ + (size_t)(rowB + v) * T_ + col] =
                    acc[ms][ns][v] * scale;
            }
        }
}

// ---------------------------------------------------------------------------
extern "C" void kernel_launch(void* const* d_in, const int* in_sizes, int n_in,
                              void* d_out, int out_size, void* d_ws, size_t ws_size,
                              hipStream_t stream) {
    const float* e = (const float*)d_in[0];      // (B,1,T) fp32
    const float* h = (const float*)d_in[1];      // (B,1,T,H) fp32
    // d_in[2] = ilens: unused by the reference output
    float* out = (float*)d_out;                  // (B,1,T,T) fp32

    float* mm  = (float*)d_ws;                             // 2 floats
    float* agg = (float*)((char*)d_ws + 256);              // B*T*H fp32 (33.5 MB)

    minmax_kernel<<<dim3(1), dim3(256), 0, stream>>>(e, mm);
    agg_scan_kernel<<<dim3(B_), dim3(512), 0, stream>>>(e, h, mm, agg);

    dim3 grid(T_ / 128, T_ / 64, B_);
    gemm_kernel<<<grid, dim3(256), 0, stream>>>(h, agg, out);
}
